// MultiHeadAttentionBlock_10995116278332
// MI455X (gfx1250) — matrix operand order repaired and checked by reading, not compiled, not run on hardware
//
#include <hip/hip_runtime.h>

// ---------------------------------------------------------------------------
// MultiHeadAttentionBlock for MI455X (gfx1250, wave32, WMMA)
//   out  = softmax((xWq)(xWk)^T / sqrt(d)) (xWv) Wout^T + b   [f32]
//   attn = the softmax matrix                                  [f32]
// bf16 operands + f32 WMMA accum; flash-style 2-pass softmax per 16-row
// q-tile (normalized attn written exactly once, NT stores); K/V^T tiles
// double-buffered into LDS with GLOBAL_LOAD_ASYNC_TO_LDS_B128 and partial
// s_wait_asynccnt waits so the async engine overlaps the WMMA stream.
// ---------------------------------------------------------------------------

#define EMBD     1024
#define NHEADS   16
#define HDIM     64
#define BATCH    2
#define SEQ      2048
#define NTOK     (BATCH * SEQ)     // 4096
#define ATTN_SCALE 0.125f          // 1/sqrt(64)

typedef __attribute__((ext_vector_type(16))) __bf16  v16bf;
typedef __attribute__((ext_vector_type(8)))  float   v8f;
typedef __attribute__((ext_vector_type(4)))  unsigned int v4u;

union FragU { v4u q[2]; v16bf v; };

#define WMMA_BF16(a, b, c) \
  __builtin_amdgcn_wmma_f32_16x16x32_bf16(false, (a), false, (b), (short)0, (c), false, false)

__device__ __forceinline__ unsigned short f32_to_bf16(float f) {
  unsigned int u = __float_as_uint(f);
  u += 0x7FFFu + ((u >> 16) & 1u);          // round-to-nearest-even
  return (unsigned short)(u >> 16);
}

// A-matrix fragment (16x32 bf16), row-major source; this code's k order (B matches):
// lane l: row m = l&15, half h = l>>4; elems e -> k = 16h+e (contiguous)
__device__ __forceinline__ v16bf load_frag_a_g(const unsigned short* base, int ld,
                                               int row0, int k0) {
  const int l = threadIdx.x & 31;
  const int m = l & 15, h = l >> 4;
  const unsigned short* p = base + (size_t)(row0 + m) * ld + (k0 + 16 * h);
  FragU f;
  f.q[0] = *reinterpret_cast<const v4u*>(p);
  f.q[1] = *reinterpret_cast<const v4u*>(p + 8);
  return f.v;
}

// B-matrix fragment (32x16 bf16); column n of B = row (col0+n) of `base`
// (contiguous in k). lane l: n = l&15, h = l>>4; elems e -> k = 16h+e (32B chunk)
__device__ __forceinline__ v16bf load_frag_b_g(const unsigned short* base, int ld,
                                               int col0, int k0) {
  const int l = threadIdx.x & 31;
  const int n = l & 15, h = l >> 4;
  const unsigned short* p = base + (size_t)(col0 + n) * ld + (k0 + 16 * h);
  FragU f;
  f.q[0] = *reinterpret_cast<const v4u*>(p);
  f.q[1] = *reinterpret_cast<const v4u*>(p + 8);
  return f.v;
}

// Cooperative 64x64-bf16 tile copy, global -> LDS, via async-to-LDS b128.
// 256 threads, 512 16-byte chunks (8 per 128B row); 2 async ops per wave.
__device__ __forceinline__ void async_copy_tile64(const unsigned short* gbase,
                                                  size_t row_stride,
                                                  unsigned short (*tile)[64],
                                                  int tid) {
#pragma unroll
  for (int it = 0; it < 2; ++it) {
    const int chunk = tid + it * 256;        // 0..511
    const int row = chunk >> 3;
    const int c16 = chunk & 7;
    const unsigned short* g = gbase + (size_t)row * row_stride + c16 * 8;
    const unsigned lds = (unsigned)(unsigned long long)(const void*)&tile[row][c16 * 8];
    const unsigned long long ga = (unsigned long long)g;
    asm volatile("global_load_async_to_lds_b128 %0, %1, off"
                 :: "v"(lds), "v"(ga) : "memory");
  }
}

// ---------------------------------------------------------------------------
// Kernel 0: f32 -> bf16 conversion (grid-stride)
// ---------------------------------------------------------------------------
__global__ void cvt_bf16_kernel(const float* __restrict__ src,
                                unsigned short* __restrict__ dst, int n) {
  for (int i = blockIdx.x * blockDim.x + threadIdx.x; i < n;
       i += gridDim.x * blockDim.x)
    dst[i] = f32_to_bf16(src[i]);
}

// ---------------------------------------------------------------------------
// Kernel 1: QKV projection.  qkv[m, n] = sum_k x[m,k] * w_qkv[n,k]
// M=4096, N=3072, K=1024.  One 64x64 tile per wave (16 WMMAs / k-step),
// operands direct from global (x: 48x reuse, w: 64x reuse -> 192MB L2 covers).
// Q,K stored [B,H,L,64] bf16; V stored transposed [B,H,64,L] bf16.
// ---------------------------------------------------------------------------
__global__ void __launch_bounds__(256)
qkv_gemm_kernel(const unsigned short* __restrict__ xb,
                const unsigned short* __restrict__ wqb,
                unsigned short* __restrict__ qbuf,
                unsigned short* __restrict__ kbuf,
                unsigned short* __restrict__ vtbuf) {
  const int wave = threadIdx.x >> 5;
  const int gt = blockIdx.x * 8 + wave;   // 0..3071
  const int mt = gt & 63;                 // 64 m-tiles
  const int nt = gt >> 6;                 // 48 n-tiles
  const int m0 = mt * 64, n0 = nt * 64;

  v8f acc[4][4] = {};
  for (int kt = 0; kt < EMBD; kt += 32) {
    v16bf a[4], b[4];
#pragma unroll
    for (int i = 0; i < 4; ++i) a[i] = load_frag_a_g(xb, EMBD, m0 + 16 * i, kt);
#pragma unroll
    for (int j = 0; j < 4; ++j) b[j] = load_frag_b_g(wqb, EMBD, n0 + 16 * j, kt);
#pragma unroll
    for (int i = 0; i < 4; ++i)
#pragma unroll
      for (int j = 0; j < 4; ++j)
        acc[i][j] = WMMA_BF16(a[i], b[j], acc[i][j]);
  }

  const int l = threadIdx.x & 31;
  const int nl = l & 15, h = l >> 4;
#pragma unroll
  for (int j = 0; j < 4; ++j) {
    // uniform per 16-col tile (tiles never straddle 1024/64 boundaries):
    const int nbase = n0 + 16 * j;
    const int sel  = nbase >> 10;              // 0=q 1=k 2=v  (scalar branch)
    const int head = (nbase & 1023) >> 6;      // scalar
    const int dd   = (nbase & 63) + nl;        // lane-varying low bits only
#pragma unroll
    for (int i = 0; i < 4; ++i) {
#pragma unroll
      for (int r = 0; r < 8; ++r) {
        const int gm = m0 + 16 * i + r + 8 * h;     // flat token
        const int bb = gm >> 11, tok = gm & 2047;
        const unsigned short bv = f32_to_bf16(acc[i][j][r]);
        const size_t bh = (size_t)(bb * NHEADS + head);
        if (sel == 0)
          qbuf[(bh * SEQ + tok) * HDIM + dd] = bv;
        else if (sel == 1)
          kbuf[(bh * SEQ + tok) * HDIM + dd] = bv;
        else
          vtbuf[(bh * HDIM + dd) * SEQ + tok] = bv;
      }
    }
  }
}

// ---------------------------------------------------------------------------
// Kernel 2: attention. Block = 8 waves = 8 q-tiles (128 rows) of ONE (b,head);
// K / V^T 64x64 tiles staged once per block into double-buffered LDS with
// async-to-LDS loads: copy of chunk c+1 overlaps WMMA+softmax of chunk c,
// synchronized with partial s_wait_asynccnt (in-order async completion).
// Pass 1: online softmax stats. Pass 2: recompute S, write normalized attn
// (NT stores, 536MB >> L2), fuse O += P*V from LDS.
// ---------------------------------------------------------------------------
__global__ void __launch_bounds__(256)
attn_kernel(const unsigned short* __restrict__ qb,
            const unsigned short* __restrict__ kb,
            const unsigned short* __restrict__ vtb,
            float* __restrict__ attn,
            unsigned short* __restrict__ ctx) {
  __shared__ __align__(16) unsigned short tileK[2][64][64];  // 16 KB
  __shared__ __align__(16) unsigned short tileV[2][64][64];  // 16 KB
  __shared__ __align__(16) unsigned short p_lds[8][16 * 64]; // 16 KB

  const int tid = threadIdx.x;
  const int wave = tid >> 5;
  const int l = tid & 31;
  const int nl = l & 15, h = l >> 4;
  const int bh = blockIdx.x >> 4;                    // b*NHEADS + head (0..31)
  const int qtile = (blockIdx.x & 15) * 8 + wave;    // 0..127
  const int row0 = qtile * 16;

  const unsigned short* Q  = qb  + (size_t)bh * SEQ * HDIM;
  const unsigned short* K  = kb  + (size_t)bh * SEQ * HDIM;
  const unsigned short* VT = vtb + (size_t)bh * HDIM * SEQ;

  const v16bf aq0 = load_frag_a_g(Q, HDIM, row0, 0);
  const v16bf aq1 = load_frag_a_g(Q, HDIM, row0, 32);

  float rmax[8], rsum[8];
#pragma unroll
  for (int r = 0; r < 8; ++r) { rmax[r] = -__builtin_inff(); rsum[r] = 0.0f; }

  // ---- Pass 1: row max / sum-of-exp (online), double-buffered K tiles ----
  async_copy_tile64(K, HDIM, tileK[0], tid);
  for (int c = 0; c < SEQ / 64; ++c) {
    const int j = c * 64;
    if (c + 1 < SEQ / 64) {
      async_copy_tile64(K + (size_t)(j + 64) * HDIM, HDIM, tileK[(c + 1) & 1], tid);
      asm volatile("s_wait_asynccnt 0x2" ::: "memory");  // chunk c done, c+1 inflight
    } else {
      asm volatile("s_wait_asynccnt 0x0" ::: "memory");
    }
    __syncthreads();
    const unsigned short* tk = &tileK[c & 1][0][0];
#pragma unroll
    for (int st = 0; st < 4; ++st) {
      v16bf b0 = load_frag_b_g(tk, 64, st * 16, 0);
      v16bf b1 = load_frag_b_g(tk, 64, st * 16, 32);
      v8f s = {};
      s = WMMA_BF16(aq0, b0, s);
      s = WMMA_BF16(aq1, b1, s);
#pragma unroll
      for (int r = 0; r < 8; ++r) {
        float v = s[r] * ATTN_SCALE;
        float tmax = v;                 // reduce over the 16 lanes of a row
        tmax = fmaxf(tmax, __shfl_xor(tmax, 1));
        tmax = fmaxf(tmax, __shfl_xor(tmax, 2));
        tmax = fmaxf(tmax, __shfl_xor(tmax, 4));
        tmax = fmaxf(tmax, __shfl_xor(tmax, 8));
        float mnew = fmaxf(rmax[r], tmax);
        float e = __expf(v - mnew);
        e += __shfl_xor(e, 1);
        e += __shfl_xor(e, 2);
        e += __shfl_xor(e, 4);
        e += __shfl_xor(e, 8);
        rsum[r] = rsum[r] * __expf(rmax[r] - mnew) + e;
        rmax[r] = mnew;
      }
    }
    __syncthreads();                   // buffer consumed; safe to overwrite next iter
  }
  float rinv[8];
#pragma unroll
  for (int r = 0; r < 8; ++r) rinv[r] = 1.0f / rsum[r];

  // ---- Pass 2: write attn, fuse O += P*V; double-buffered K and V tiles ----
  v8f o[4] = {};
  float* attn_bh = attn + (size_t)bh * SEQ * SEQ;
  async_copy_tile64(K, HDIM, tileK[0], tid);
  async_copy_tile64(VT, SEQ, tileV[0], tid);
  for (int c = 0; c < SEQ / 64; ++c) {
    const int j = c * 64;
    if (c + 1 < SEQ / 64) {
      async_copy_tile64(K + (size_t)(j + 64) * HDIM, HDIM, tileK[(c + 1) & 1], tid);
      async_copy_tile64(VT + (j + 64), SEQ, tileV[(c + 1) & 1], tid);
      asm volatile("s_wait_asynccnt 0x4" ::: "memory");  // chunk c done, c+1 inflight
    } else {
      asm volatile("s_wait_asynccnt 0x0" ::: "memory");
    }
    __syncthreads();
    const unsigned short* tk = &tileK[c & 1][0][0];
    const unsigned short* tv = &tileV[c & 1][0][0];
#pragma unroll
    for (int st = 0; st < 4; ++st) {
      v16bf b0 = load_frag_b_g(tk, 64, st * 16, 0);
      v16bf b1 = load_frag_b_g(tk, 64, st * 16, 32);
      v8f s = {};
      s = WMMA_BF16(aq0, b0, s);
      s = WMMA_BF16(aq1, b1, s);
#pragma unroll
      for (int r = 0; r < 8; ++r) {
        const int m = r + 8 * h;
        const float p = __expf(s[r] * ATTN_SCALE - rmax[r]) * rinv[r];
        // stream normalized attn out with non-temporal stores
        __builtin_nontemporal_store(
            p, attn_bh + (size_t)(row0 + m) * SEQ + j + st * 16 + nl);
        p_lds[wave][m * 64 + st * 16 + nl] = f32_to_bf16(p);
      }
    }
    // D-layout -> A-layout transpose of P through per-wave LDS tile
    asm volatile("s_wait_dscnt 0" ::: "memory");
    __builtin_amdgcn_wave_barrier();
    const v16bf pa0 = load_frag_a_g(&p_lds[wave][0], 64, 0, 0);
    const v16bf pa1 = load_frag_a_g(&p_lds[wave][0], 64, 0, 32);
    __builtin_amdgcn_wave_barrier();
#pragma unroll
    for (int t = 0; t < 4; ++t) {
      v16bf bv0 = load_frag_b_g(tv, 64, 16 * t, 0);
      v16bf bv1 = load_frag_b_g(tv, 64, 16 * t, 32);
      o[t] = WMMA_BF16(pa0, bv0, o[t]);
      o[t] = WMMA_BF16(pa1, bv1, o[t]);
    }
    __syncthreads();                   // buffers consumed; safe to overwrite
  }

  // context (already normalized): ctx[b, tok, head*64 + dd] bf16
  const int head = bh & 15, bb = bh >> 4;
#pragma unroll
  for (int t = 0; t < 4; ++t)
#pragma unroll
    for (int r = 0; r < 8; ++r) {
      const int tok = row0 + r + 8 * h;
      ctx[((size_t)bb * SEQ + tok) * EMBD + head * HDIM + 16 * t + nl] =
          f32_to_bf16(o[t][r]);
    }
}

// ---------------------------------------------------------------------------
// Kernel 3: output projection. out[m,n] = sum_k ctx[m,k]*w_out[n,k] + b_out[n]
// M=4096, N=1024, K=1024; 64x64 tile per wave, f32 output.
// ---------------------------------------------------------------------------
__global__ void __launch_bounds__(256)
out_proj_kernel(const unsigned short* __restrict__ ctx,
                const unsigned short* __restrict__ wob,
                const float* __restrict__ b_out,
                float* __restrict__ out) {
  const int wave = threadIdx.x >> 5;
  const int gt = blockIdx.x * 8 + wave;   // 0..1023
  const int mt = gt & 63, nt = gt >> 6;   // 64 x 16 tiles
  const int m0 = mt * 64, n0 = nt * 64;

  v8f acc[4][4] = {};
  for (int kt = 0; kt < EMBD; kt += 32) {
    v16bf a[4], b[4];
#pragma unroll
    for (int i = 0; i < 4; ++i) a[i] = load_frag_a_g(ctx, EMBD, m0 + 16 * i, kt);
#pragma unroll
    for (int j = 0; j < 4; ++j) b[j] = load_frag_b_g(wob, EMBD, n0 + 16 * j, kt);
#pragma unroll
    for (int i = 0; i < 4; ++i)
#pragma unroll
      for (int j = 0; j < 4; ++j)
        acc[i][j] = WMMA_BF16(a[i], b[j], acc[i][j]);
  }

  const int l = threadIdx.x & 31;
  const int nl = l & 15, h = l >> 4;
#pragma unroll
  for (int j = 0; j < 4; ++j) {
    const int gn = n0 + 16 * j + nl;
    const float bias = b_out[gn];
#pragma unroll
    for (int i = 0; i < 4; ++i)
#pragma unroll
      for (int r = 0; r < 8; ++r) {
        const int gm = m0 + 16 * i + r + 8 * h;
        out[(size_t)gm * EMBD + gn] = acc[i][j][r] + bias;
      }
  }
}

// ---------------------------------------------------------------------------
// Host-side launch
// ---------------------------------------------------------------------------
extern "C" void kernel_launch(void* const* d_in, const int* in_sizes, int n_in,
                              void* d_out, int out_size, void* d_ws, size_t ws_size,
                              hipStream_t stream) {
  (void)in_sizes; (void)n_in; (void)out_size; (void)ws_size;

  const float* x     = (const float*)d_in[0];   // [2,2048,1024]
  const float* w_qkv = (const float*)d_in[1];   // [3072,1024]
  const float* w_out = (const float*)d_in[2];   // [1024,1024]
  const float* b_out = (const float*)d_in[3];   // [1024]

  float* out  = (float*)d_out;                       // [2,2048,1024]
  float* attn = (float*)d_out + (size_t)NTOK * EMBD; // [2,16,2048,2048]

  // Workspace layout (u16 elements): 48 MiB total
  unsigned short* ws   = (unsigned short*)d_ws;
  unsigned short* xb   = ws;                    // 4,194,304  x bf16
  unsigned short* wqb  = xb  + 4194304;         // 3,145,728  w_qkv bf16
  unsigned short* wob  = wqb + 3145728;         // 1,048,576  w_out bf16
  unsigned short* qb   = wob + 1048576;         // 4,194,304  Q  [B,H,L,64]
  unsigned short* kb   = qb  + 4194304;         // 4,194,304  K  [B,H,L,64]
  unsigned short* vtb  = kb  + 4194304;         // 4,194,304  V^T[B,H,64,L]
  unsigned short* ctx  = vtb + 4194304;         // 4,194,304  context [B,L,E]

  cvt_bf16_kernel<<<2048, 256, 0, stream>>>(x, xb, NTOK * EMBD);
  cvt_bf16_kernel<<<2048, 256, 0, stream>>>(w_qkv, wqb, 3 * EMBD * EMBD);
  cvt_bf16_kernel<<<1024, 256, 0, stream>>>(w_out, wob, EMBD * EMBD);

  // 64x64 tiles: (4096/64)*(3072/64)=3072 wave-tiles, 8 waves/block
  qkv_gemm_kernel<<<384, 256, 0, stream>>>(xb, wqb, qb, kb, vtb);

  // 32 (b,head) x 16 blocks, each block = 8 q-tiles of 16 rows
  attn_kernel<<<512, 256, 0, stream>>>(qb, kb, vtb, attn, ctx);

  // (4096/64)*(1024/64) = 1024 wave-tiles
  out_proj_kernel<<<128, 256, 0, stream>>>(ctx, wob, b_out, out);
}
